// SelfAttention_24154896073280
// MI455X (gfx1250) — compile-verified, matrix operand order checked by reading, not run on hardware
//
#include <hip/hip_runtime.h>

#define NT    8192
#define DIN   512
#define DOUT  64
#define SPLIT 4
#define NCHUNK (NT / SPLIT)   // keys per split = 2048

typedef __attribute__((ext_vector_type(16))) _Float16 v16h;
typedef __attribute__((ext_vector_type(8)))  _Float16 v8h;
typedef __attribute__((ext_vector_type(8)))  float    v8f;

static __device__ __forceinline__ v8f wmma_f16(v16h a, v16h b, v8f c) {
  // 8 args: (neg_a, A, neg_b, B, c_mod, C, reuse_a, reuse_b)
  return __builtin_amdgcn_wmma_f32_16x16x32_f16(false, a, false, b, (short)0, c,
                                                false, false);
}

// 16-byte async DMA: global -> LDS, tracked by ASYNCcnt (ISA §15.18.3 op 98).
// Generic pointers to __shared__ carry the LDS byte offset in their low 32 bits.
static __device__ __forceinline__ void async_cp16(const void* gsrc, void* ldst) {
  unsigned laddr = (unsigned)(size_t)ldst;
  asm volatile("global_load_async_to_lds_b128 %0, %1, off"
               :: "v"(laddr), "v"(gsrc) : "memory");
}
#define WAIT_ASYNC() asm volatile("s_wait_asynccnt 0x0" ::: "memory")

// ---------------------------------------------------------------------------
// Kernel 0: one-time f32 -> f16 conversion (input + weights), vectorized x8.
// ---------------------------------------------------------------------------
__global__ __launch_bounds__(256) void cvt_f16_kernel(
    const float* __restrict__ src, _Float16* __restrict__ dst, int n) {
  int i = (blockIdx.x * 256 + threadIdx.x) * 8;
  if (i < n) {
    v8f x = *(const v8f*)(src + i);
    *(v8h*)(dst + i) = __builtin_convertvector(x, v8h);
  }
}

// ---------------------------------------------------------------------------
// Phase 1: QKV projection, pure-f16 WMMA hot loop.
//   grid = (NT/16, 3)   blockIdx.y: 0=Q, 1=K, 2=V
// Q,K -> row-major f16 [NT][64];  V -> transposed f16 [64][NT].
// ---------------------------------------------------------------------------
__global__ __launch_bounds__(32) void qkv_proj_kernel(
    const _Float16* __restrict__ inH,   // [NT][DIN] f16
    const _Float16* __restrict__ wH,    // [3][DOUT][DIN] f16
    const float* __restrict__ bq, const float* __restrict__ bk,
    const float* __restrict__ bv,
    _Float16* __restrict__ qh, _Float16* __restrict__ kh,
    _Float16* __restrict__ vT)
{
  const int lane    = threadIdx.x & 31;
  const int r       = lane & 15;   // row (A) / column (B) index
  const int h       = lane >> 4;   // K-half selector
  const int rowBase = blockIdx.x * 16;
  const int which   = blockIdx.y;

  const _Float16* W  = wH + (size_t)which * DOUT * DIN;
  const float*    bi = (which == 0) ? bq : (which == 1) ? bk : bv;

  v8f acc[4] = {};
  for (int kb = 0; kb < DIN; kb += 32) {
    v16h a = *(const v16h*)(inH + (size_t)(rowBase + r) * DIN + kb + h * 16);
#pragma unroll
    for (int t = 0; t < 4; ++t) {
      v16h b = *(const v16h*)(W + (size_t)(t * 16 + r) * DIN + kb + h * 16);
      acc[t] = wmma_f16(a, b, acc[t]);
    }
  }

  // D layout: lane half h, VGPR j -> row (h*8+j); lane&15 -> column
  if (which < 2) {
    _Float16* dst = (which == 0) ? qh : kh;
#pragma unroll
    for (int t = 0; t < 4; ++t) {
      float bb = bi[t * 16 + r];
#pragma unroll
      for (int j = 0; j < 8; ++j)
        dst[(size_t)(rowBase + h * 8 + j) * DOUT + t * 16 + r] =
            (_Float16)(acc[t][j] + bb);
    }
  } else {
    // transposed store: fixed dim per lane, 8 consecutive rows -> 16B vector
#pragma unroll
    for (int t = 0; t < 4; ++t) {
      float bb = bi[t * 16 + r];
      v8h pack;
#pragma unroll
      for (int j = 0; j < 8; ++j) pack[j] = (_Float16)(acc[t][j] + bb);
      *(v8h*)(vT + (size_t)(t * 16 + r) * NT + rowBase + h * 8) = pack;
    }
  }
}

// ---------------------------------------------------------------------------
// Phase 2: split-K flash attention.  grid = (NT/64, SPLIT), block = 128.
// Each wave owns 16 query rows and one key range of NCHUNK keys; emits
// partial (O_s, m_s, l_s).  K/V tiles staged block-cooperatively into
// double-buffered LDS with global_load_async_to_lds_b128.
// ---------------------------------------------------------------------------
__global__ __launch_bounds__(128) void flash_attn_kernel(
    const _Float16* __restrict__ qh, const _Float16* __restrict__ kh,
    const _Float16* __restrict__ vT,
    float* __restrict__ po,   // [SPLIT][NT][DOUT]
    float* __restrict__ pm,   // [SPLIT][NT]
    float* __restrict__ pl)   // [SPLIT][NT]
{
  __shared__ alignas(128) _Float16 Ks[2][32][DOUT];  // 4 KB per buffer
  __shared__ alignas(128) _Float16 Vs[2][DOUT][32];  // 4 KB per buffer
  __shared__ alignas(64)  _Float16 Pbuf[4][16][32];  // per-wave P transpose

  const int tid   = threadIdx.x;
  const int wave  = tid >> 5;
  const int lane  = tid & 31;
  const int r     = lane & 15;
  const int h     = lane >> 4;
  const int qBase = (blockIdx.x * 4 + wave) * 16;
  const int split = blockIdx.y;
  const int k0    = split * NCHUNK;

  // Q fragments (A operand), contraction over d=64 -> two K=32 chunks
  v16h q0 = *(const v16h*)(qh + (size_t)(qBase + r) * DOUT + h * 16);
  v16h q1 = *(const v16h*)(qh + (size_t)(qBase + r) * DOUT + 32 + h * 16);

  float m_i[8], l_i[8];
#pragma unroll
  for (int j = 0; j < 8; ++j) { m_i[j] = -3.0e38f; l_i[j] = 0.0f; }
  v8f o[4] = {};

  // cooperative async stage of one 32-key block into LDS buffer `buf`
  auto stage = [&](int buf, int kb) {
    // K tile: 32 rows x 128B, contiguous 4 KB region of kh
    const _Float16* kg = kh + (size_t)kb * DOUT;
    async_cp16(kg + tid * 8,        &Ks[buf][0][0] + tid * 8);
    async_cp16(kg + 1024 + tid * 8, &Ks[buf][0][0] + 1024 + tid * 8);
    // V tile: 64 rows of 64B gathered from vT (row stride NT); chunk = 16B
    int c0 = tid, c1 = tid + 128;
    async_cp16(vT + (size_t)(c0 >> 2) * NT + kb + (c0 & 3) * 8,
               &Vs[buf][0][0] + c0 * 8);
    async_cp16(vT + (size_t)(c1 >> 2) * NT + kb + (c1 & 3) * 8,
               &Vs[buf][0][0] + c1 * 8);
  };

  stage(0, k0);
  WAIT_ASYNC();
  __syncthreads();

  for (int kb = k0; kb < k0 + NCHUNK; kb += 32) {
    const int buf = ((kb - k0) >> 5) & 1;
    if (kb + 32 < k0 + NCHUNK) stage(buf ^ 1, kb + 32);  // overlap next tile
    if (kb + 64 < k0 + NCHUNK)                           // deep L2 prefetch
      __builtin_prefetch(kh + (size_t)(kb + 64) * DOUT + lane * 64, 0, 1);

    // S = Q @ K^T for two 16-key sub-tiles (B columns are K rows, from LDS)
    v8f s[2];
#pragma unroll
    for (int c = 0; c < 2; ++c) {
      v16h b0 = *(const v16h*)&Ks[buf][c * 16 + r][h * 16];
      v16h b1 = *(const v16h*)&Ks[buf][c * 16 + r][32 + h * 16];
      v8f sc = {};
      sc = wmma_f16(q0, b0, sc);
      sc = wmma_f16(q1, b1, sc);
      s[c] = sc;
    }

    // Online softmax; row (h*8+j) is spread over the 16 lanes of this half.
    float p0[8], p1[8], scal[8];
#pragma unroll
    for (int j = 0; j < 8; ++j) {
      float mn = fmaxf(m_i[j], fmaxf(s[0][j], s[1][j]));
#pragma unroll
      for (int mask = 1; mask <= 8; mask <<= 1)
        mn = fmaxf(mn, __shfl_xor(mn, mask));
      scal[j] = __expf(m_i[j] - mn);
      p0[j]   = __expf(s[0][j] - mn);
      p1[j]   = __expf(s[1][j] - mn);
      float rs = p0[j] + p1[j];
#pragma unroll
      for (int mask = 1; mask <= 8; mask <<= 1)
        rs += __shfl_xor(rs, mask);
      l_i[j] = l_i[j] * scal[j] + rs;
      m_i[j] = mn;
    }
#pragma unroll
    for (int t = 0; t < 4; ++t)
#pragma unroll
      for (int j = 0; j < 8; ++j) o[t][j] *= scal[j];

    // D-layout (key-per-lane) -> A-layout (row-per-lane) via LDS transpose
    __syncthreads();
#pragma unroll
    for (int j = 0; j < 8; ++j) {
      Pbuf[wave][h * 8 + j][r]      = (_Float16)p0[j];
      Pbuf[wave][h * 8 + j][16 + r] = (_Float16)p1[j];
    }
    __syncthreads();

    v16h pa = *(const v16h*)&Pbuf[wave][r][h * 16];
#pragma unroll
    for (int t = 0; t < 4; ++t) {
      v16h vb = *(const v16h*)&Vs[buf][t * 16 + r][h * 16];
      o[t] = wmma_f16(pa, vb, o[t]);
    }

    WAIT_ASYNC();        // next buffer fully landed in LDS
    __syncthreads();     // ...for every wave in the block
  }

  // partial output (unnormalized) + per-row stats
  float* od = po + (size_t)split * NT * DOUT;
#pragma unroll
  for (int t = 0; t < 4; ++t)
#pragma unroll
    for (int j = 0; j < 8; ++j)
      od[(size_t)(qBase + h * 8 + j) * DOUT + t * 16 + r] = o[t][j];
  if (r == 0) {
#pragma unroll
    for (int j = 0; j < 8; ++j) {
      pm[(size_t)split * NT + qBase + h * 8 + j] = m_i[j];
      pl[(size_t)split * NT + qBase + h * 8 + j] = l_i[j];
    }
  }
}

// ---------------------------------------------------------------------------
// Phase 3: combine split-K partials.  out = sum_s O_s e^{m_s-m*} / sum_s l_s e^{m_s-m*}
// ---------------------------------------------------------------------------
__global__ __launch_bounds__(256) void combine_kernel(
    const float* __restrict__ po, const float* __restrict__ pm,
    const float* __restrict__ pl, float* __restrict__ out)
{
  int idx = blockIdx.x * 256 + threadIdx.x;   // row * DOUT + dim
  int row = idx >> 6;
  float m = -3.0e38f;
#pragma unroll
  for (int s = 0; s < SPLIT; ++s) m = fmaxf(m, pm[(size_t)s * NT + row]);
  float num = 0.0f, den = 0.0f;
#pragma unroll
  for (int s = 0; s < SPLIT; ++s) {
    float w = __expf(pm[(size_t)s * NT + row] - m);
    num += po[(size_t)s * NT * DOUT + idx] * w;
    den += pl[(size_t)s * NT + row] * w;
  }
  out[idx] = num / den;
}

// ---------------------------------------------------------------------------
extern "C" void kernel_launch(void* const* d_in, const int* in_sizes, int n_in,
                              void* d_out, int out_size, void* d_ws,
                              size_t ws_size, hipStream_t stream) {
  (void)in_sizes; (void)n_in; (void)out_size; (void)ws_size;

  const float* input = (const float*)d_in[0];
  const float* Wq = (const float*)d_in[1];
  const float* bq = (const float*)d_in[2];
  const float* Wk = (const float*)d_in[3];
  const float* bk = (const float*)d_in[4];
  const float* Wv = (const float*)d_in[5];
  const float* bv = (const float*)d_in[6];
  float* out = (float*)d_out;

  // ws: inH 8M | wH 256K | qh 1M | kh 1M | vT 1M | po 8M | pm 128K | pl 128K
  char* ws = (char*)d_ws;
  _Float16* inH = (_Float16*)ws;
  _Float16* wH  = (_Float16*)(ws + (8u << 20));
  _Float16* qh  = (_Float16*)(ws + (8u << 20) + (1u << 18));
  _Float16* kh  = qh + (size_t)NT * DOUT;
  _Float16* vT  = kh + (size_t)NT * DOUT;
  float*    po  = (float*)(vT + (size_t)NT * DOUT);
  float*    pm  = po + (size_t)SPLIT * NT * DOUT;
  float*    pl  = pm + (size_t)SPLIT * NT;

  cvt_f16_kernel<<<NT * DIN / (256 * 8), 256, 0, stream>>>(input, inH, NT * DIN);
  cvt_f16_kernel<<<DOUT * DIN / (256 * 8), 256, 0, stream>>>(Wq, wH, DOUT * DIN);
  cvt_f16_kernel<<<DOUT * DIN / (256 * 8), 256, 0, stream>>>(
      Wk, wH + (size_t)DOUT * DIN, DOUT * DIN);
  cvt_f16_kernel<<<DOUT * DIN / (256 * 8), 256, 0, stream>>>(
      Wv, wH + (size_t)2 * DOUT * DIN, DOUT * DIN);

  qkv_proj_kernel<<<dim3(NT / 16, 3), 32, 0, stream>>>(inH, wH, bq, bk, bv, qh,
                                                       kh, vT);
  flash_attn_kernel<<<dim3(NT / 64, SPLIT), 128, 0, stream>>>(qh, kh, vT, po,
                                                              pm, pl);
  combine_kernel<<<NT * DOUT / 256, 256, 0, stream>>>(po, pm, pl, out);
}
